// Attention_59236188947093
// MI455X (gfx1250) — compile-verified, matrix operand order checked by reading, not run on hardware
//
#include <hip/hip_runtime.h>
#include <hip/hip_bf16.h>

// ---------------------------------------------------------------------------
// Fused attention for MI455X (gfx1250, wave32, WMMA, async-LDS copies).
//   x[4,8,512,512] f32, pos_bias (softmax-invariant -> unused),
//   w_qkv[512,1536] f32, w_out[512,512] f32  ->  out[4,8,512,512] f32
//
// Pipeline:
//   0) x -> f16; w_qkv,w_out -> transposed f16  (bandwidth-trivial)
//   1) qkv = x @ W_qkv        : WMMA GEMM, 128x128 tile, async double-buffer
//   2) flash attention         : 64-wide KV tiles, online softmax, row-sums
//                                folded into an extra WMMA vs an all-ones frag
//   3) out = attn @ W_out      : same GEMM, f32 output
// ~52 GFLOP of matrix math vs ~37 MB HBM traffic -> matrix-op bound; all
// matmuls go through v_wmma_f32_16x16x32_f16, intermediates live in L2.
// ---------------------------------------------------------------------------

typedef _Float16 v16h __attribute__((ext_vector_type(16)));
typedef _Float16 v8h  __attribute__((ext_vector_type(8)));
typedef float    v8f  __attribute__((ext_vector_type(8)));
typedef int      v4i  __attribute__((ext_vector_type(4)));

#define WMMA_F32_F16(a, b, c) \
  __builtin_amdgcn_wmma_f32_16x16x32_f16(false, (a), false, (b), (short)0, (c), false, false)

// ---- CDNA5 async global->LDS copy (ASYNCcnt), with sync fallback ----------
#if __has_builtin(__builtin_amdgcn_global_load_async_to_lds_b128) && \
    __has_builtin(__builtin_amdgcn_s_wait_asynccnt)
#define HAS_ASYNC_LDS 1
#else
#define HAS_ASYNC_LDS 0
#endif

// pointer-to-AS(1)/AS(3) int4, matching the builtin's parameter types
// (clang prints AS(1) as "__device__", AS(3) as "__shared__" in HIP mode)
typedef __attribute__((address_space(1))) v4i* gptr_b128;
typedef __attribute__((address_space(3))) v4i* lptr_b128;

__device__ __forceinline__ void copy16(const void* g, void* l) {
#if HAS_ASYNC_LDS
  __builtin_amdgcn_global_load_async_to_lds_b128(
      (gptr_b128)(uintptr_t)g, (lptr_b128)(uint32_t)(uintptr_t)l, 0, 0);
#else
  *(uint4*)l = *(const uint4*)g;
#endif
}
__device__ __forceinline__ void async_join() {
#if HAS_ASYNC_LDS
  __builtin_amdgcn_s_wait_asynccnt(0);
#endif
}

// ---------------------------------------------------------------------------
// 0a) f32 -> f16 elementwise (8 elems / thread, 16B stores)
// ---------------------------------------------------------------------------
__global__ __launch_bounds__(256) void cvt_f16_kernel(
    const float* __restrict__ in, _Float16* __restrict__ out, int n) {
  int i = (blockIdx.x * 256 + threadIdx.x) * 8;
  if (i < n) {
    float4 a = *(const float4*)(in + i);
    float4 b = *(const float4*)(in + i + 4);
    v8h o;
    o[0] = (_Float16)a.x; o[1] = (_Float16)a.y;
    o[2] = (_Float16)a.z; o[3] = (_Float16)a.w;
    o[4] = (_Float16)b.x; o[5] = (_Float16)b.y;
    o[6] = (_Float16)b.z; o[7] = (_Float16)b.w;
    *(v8h*)(out + i) = o;
  }
}

// ---------------------------------------------------------------------------
// 0b) transpose + convert: out[n][k] = (f16) in[k][n], 32x32 LDS tiles
// ---------------------------------------------------------------------------
__global__ __launch_bounds__(256) void transpose_f16_kernel(
    const float* __restrict__ in, _Float16* __restrict__ out, int K, int N) {
  __shared__ float t[32][33];
  const int k0 = blockIdx.y * 32, n0 = blockIdx.x * 32;
  const int tx = threadIdx.x & 31, ty = threadIdx.x >> 5;  // ty 0..7
#pragma unroll
  for (int i = 0; i < 4; ++i)
    t[ty + i * 8][tx] = in[(size_t)(k0 + ty + i * 8) * N + n0 + tx];
  __syncthreads();
#pragma unroll
  for (int i = 0; i < 4; ++i)
    out[(size_t)(n0 + ty + i * 8) * K + k0 + tx] = (_Float16)t[tx][ty + i * 8];
}

// ---------------------------------------------------------------------------
// 1/3) WMMA GEMM:  C[M,N] = A[M,K] * BT[N,K]^T   (both operands f16, K-major)
//   block = 256 threads (8 waves), tile 128x128, K-step 32, double-buffered
//   async LDS staging. Wave = 32x64 sub-tile = 2x4 WMMAs per K-step.
// ---------------------------------------------------------------------------
template <bool OUT_HALF>
__global__ __launch_bounds__(256) void wmma_gemm_f16(
    const _Float16* __restrict__ A, const _Float16* __restrict__ BT,
    void* __restrict__ Cv, int K, int lda, int ldbt, int ldc) {
  constexpr int LDT = 40;  // 32 + 8 pad halves: 80 B rows, conflict-free frags
  __shared__ _Float16 At[2][128][LDT];
  __shared__ _Float16 Bt[2][128][LDT];

  const int tid  = threadIdx.x;
  const int lane = tid & 31;
  const int wave = tid >> 5;          // 0..7
  const int wm   = (wave >> 1) * 32;  // 4 M groups of 32
  const int wn   = (wave & 1) * 64;   // 2 N groups of 64
  const int row0 = blockIdx.y * 128;
  const int col0 = blockIdx.x * 128;

  const int fr   = lane & 15;
  const int koff = (lane >> 4) * 8;
  const int hi   = lane >> 4;

  v8f acc[2][4] = {};

  auto stage = [&](int p, int k0) {
#pragma unroll
    for (int i = 0; i < 2; ++i) {
      int e = tid + i * 256;            // 0..511 : 128 rows x 4 16B chunks
      int r = e >> 2, c8 = (e & 3) * 8;
      copy16(A + (size_t)(row0 + r) * lda + k0 + c8, &At[p][r][c8]);
      copy16(BT + (size_t)(col0 + r) * ldbt + k0 + c8, &Bt[p][r][c8]);
    }
  };

  stage(0, 0);
  async_join();
  __syncthreads();

  const int NT = K / 32;
  for (int kt = 0; kt < NT; ++kt) {
    const int p = kt & 1;
    if (kt + 1 < NT) stage(p ^ 1, (kt + 1) * 32);  // overlap with compute

    v16h a[2], b[4];
#pragma unroll
    for (int t = 0; t < 2; ++t)
#pragma unroll
      for (int v = 0; v < 8; ++v) {
        int kb = (v & 3) * 2 + (v >> 2) * 16 + koff;
        a[t][2 * v]     = At[p][wm + t * 16 + fr][kb];
        a[t][2 * v + 1] = At[p][wm + t * 16 + fr][kb + 1];
      }
#pragma unroll
    for (int u = 0; u < 4; ++u)
#pragma unroll
      for (int v = 0; v < 8; ++v) {
        int kb = (v & 3) * 2 + (v >> 2) * 16 + koff;
        b[u][2 * v]     = Bt[p][wn + u * 16 + fr][kb];
        b[u][2 * v + 1] = Bt[p][wn + u * 16 + fr][kb + 1];
      }
#pragma unroll
    for (int tm = 0; tm < 2; ++tm)
#pragma unroll
      for (int tn = 0; tn < 4; ++tn)
        acc[tm][tn] = WMMA_F32_F16(a[tm], b[tn], acc[tm][tn]);

    async_join();
    __syncthreads();
  }

#pragma unroll
  for (int tm = 0; tm < 2; ++tm)
#pragma unroll
    for (int tn = 0; tn < 4; ++tn)
#pragma unroll
      for (int c = 0; c < 8; ++c) {
        size_t r   = (size_t)(row0 + wm + tm * 16 + c + 8 * hi);
        size_t col = (size_t)(col0 + wn + tn * 16 + fr);
        if constexpr (OUT_HALF)
          ((_Float16*)Cv)[r * ldc + col] = (_Float16)acc[tm][tn][c];
        else
          ((float*)Cv)[r * ldc + col] = acc[tm][tn][c];
      }
}

// ---------------------------------------------------------------------------
// 2) Flash attention: block = (bm, head, 128-row q slab); 8 waves x 16 q rows.
//    KV tiles 64x64, double-buffered async staging. Row-sums of P folded into
//    WMMA vs all-ones fragment (l rescales by alpha exactly like O).
//    pos_bias omitted: per-query-row constant -> softmax-invariant.
// ---------------------------------------------------------------------------
__global__ __launch_bounds__(256) void flash_attn_kernel(
    const _Float16* __restrict__ qkv,  // [16384 x 1536] f16, q|k|v
    _Float16* __restrict__ out) {      // [16384 x 512] f16
  const int blk = blockIdx.x;
  const int rb  = blk & 3;
  const int bmh = blk >> 2;
  const int h   = bmh & 7;
  const int bm  = bmh >> 3;

  const int tid  = threadIdx.x;
  const int lane = tid & 31;
  const int wave = tid >> 5;
  const int i0   = rb * 128 + wave * 16;

  const size_t ldq = 1536;
  const _Float16* qbase = qkv + (size_t)bm * 512 * ldq + h * 64;
  const _Float16* kbase = qbase + 512;
  const _Float16* vbase = qbase + 1024;

  __shared__ _Float16 Kt[2][64][72];
  __shared__ _Float16 Vt[2][64][72];
  __shared__ _Float16 Pt[8][16][72];  // wave-private P scratch

  const int fr   = lane & 15;
  const int koff = (lane >> 4) * 8;
  const int hi   = lane >> 4;

  // Q fragments (16 x 64 -> 2 frags over K)
  v16h qa[2];
#pragma unroll
  for (int t = 0; t < 2; ++t) {
    const _Float16* qr = qbase + (size_t)(i0 + fr) * ldq + t * 32;
#pragma unroll
    for (int v = 0; v < 8; ++v) {
      int kb = (v & 3) * 2 + (v >> 2) * 16 + koff;
      qa[t][2 * v]     = qr[kb];
      qa[t][2 * v + 1] = qr[kb + 1];
    }
  }

  v16h ones;
#pragma unroll
  for (int i = 0; i < 16; ++i) ones[i] = (_Float16)1.0f;

  v8f o[4] = {};
  v8f lacc = {};
  float mrun[8];
#pragma unroll
  for (int c = 0; c < 8; ++c) mrun[c] = -1e30f;
  const float scale = 0.125f;  // 64^-0.5

  auto stageKV = [&](int p, int j0) {
#pragma unroll
    for (int i = 0; i < 2; ++i) {
      int e = tid + i * 256;            // 0..511 : 64 rows x 8 16B chunks
      int r = e >> 3, c8 = (e & 7) * 8;
      copy16(kbase + (size_t)(j0 + r) * ldq + c8, &Kt[p][r][c8]);
      copy16(vbase + (size_t)(j0 + r) * ldq + c8, &Vt[p][r][c8]);
    }
  };

  stageKV(0, 0);
  async_join();
  __syncthreads();

  for (int jt = 0; jt < 8; ++jt) {     // 8 KV tiles of 64
    const int p = jt & 1;
    if (jt + 1 < 8) stageKV(p ^ 1, (jt + 1) * 64);

    // ---- S = Q K^T (16 x 64) ----
    v8f s[4];
#pragma unroll
    for (int nb = 0; nb < 4; ++nb) {
      s[nb] = v8f{};
#pragma unroll
      for (int t = 0; t < 2; ++t) {
        v16h bk;
#pragma unroll
        for (int v = 0; v < 8; ++v) {
          int kb = (v & 3) * 2 + (v >> 2) * 16 + koff;
          bk[2 * v]     = Kt[p][nb * 16 + fr][t * 32 + kb];
          bk[2 * v + 1] = Kt[p][nb * 16 + fr][t * 32 + kb + 1];
        }
        s[nb] = WMMA_F32_F16(qa[t], bk, s[nb]);
      }
    }

    // ---- online softmax (row-max only; row-sum via ones-WMMA below) ----
#pragma unroll
    for (int c = 0; c < 8; ++c) {
      float v0 = s[0][c] * scale, v1 = s[1][c] * scale;
      float v2 = s[2][c] * scale, v3 = s[3][c] * scale;
      float rm = fmaxf(fmaxf(v0, v1), fmaxf(v2, v3));
#pragma unroll
      for (int m = 1; m <= 8; m <<= 1) rm = fmaxf(rm, __shfl_xor(rm, m, 32));
      float mnew  = fmaxf(mrun[c], rm);
      float alpha = __expf(mrun[c] - mnew);
      mrun[c] = mnew;
      Pt[wave][c + 8 * hi][0 * 16 + fr]  = (_Float16)__expf(v0 - mnew);
      Pt[wave][c + 8 * hi][1 * 16 + fr]  = (_Float16)__expf(v1 - mnew);
      Pt[wave][c + 8 * hi][2 * 16 + fr]  = (_Float16)__expf(v2 - mnew);
      Pt[wave][c + 8 * hi][3 * 16 + fr]  = (_Float16)__expf(v3 - mnew);
      lacc[c] *= alpha;
#pragma unroll
      for (int d = 0; d < 4; ++d) o[d][c] *= alpha;
    }
    asm volatile("s_wait_dscnt 0" ::: "memory");  // in-wave LDS RAW fence

    // ---- P fragments (16 x 64) ----
    v16h pa[2];
#pragma unroll
    for (int t = 0; t < 2; ++t)
#pragma unroll
      for (int v = 0; v < 8; ++v) {
        int kb = (v & 3) * 2 + (v >> 2) * 16 + koff;
        pa[t][2 * v]     = Pt[wave][fr][t * 32 + kb];
        pa[t][2 * v + 1] = Pt[wave][fr][t * 32 + kb + 1];
      }

    // ---- l += rowsum(P) : WMMA against all-ones fragment ----
#pragma unroll
    for (int t = 0; t < 2; ++t) lacc = WMMA_F32_F16(pa[t], ones, lacc);

    // ---- O += P V ----
#pragma unroll
    for (int db = 0; db < 4; ++db) {
#pragma unroll
      for (int t = 0; t < 2; ++t) {
        v16h bv;
#pragma unroll
        for (int v = 0; v < 8; ++v) {
          int kb = (v & 3) * 2 + (v >> 2) * 16 + koff;
          bv[2 * v]     = Vt[p][t * 32 + kb][db * 16 + fr];
          bv[2 * v + 1] = Vt[p][t * 32 + kb + 1][db * 16 + fr];
        }
        o[db] = WMMA_F32_F16(pa[t], bv, o[db]);
      }
    }

    async_join();
    __syncthreads();
  }

  // ---- normalize and store ----
#pragma unroll
  for (int c = 0; c < 8; ++c) {
    float inv = 1.0f / lacc[c];   // every lane holds its row's sum
    size_t row = (size_t)bm * 512 + i0 + c + 8 * hi;
#pragma unroll
    for (int db = 0; db < 4; ++db)
      out[row * 512 + h * 64 + db * 16 + fr] = (_Float16)(o[db][c] * inv);
  }
}

// ---------------------------------------------------------------------------
extern "C" void kernel_launch(void* const* d_in, const int* in_sizes, int n_in,
                              void* d_out, int out_size, void* d_ws,
                              size_t ws_size, hipStream_t stream) {
  (void)in_sizes; (void)n_in; (void)out_size; (void)ws_size;
  const float* x     = (const float*)d_in[0];
  // d_in[1] = pos_bias: per-query-row constant -> softmax-invariant; unused.
  const float* w_qkv = (const float*)d_in[2];
  const float* w_out = (const float*)d_in[3];
  float* out = (float*)d_out;

  // workspace layout (bytes)
  char* ws = (char*)d_ws;
  _Float16* x_h    = (_Float16*)(ws);                        // 16384x512
  _Float16* qkv_h  = (_Float16*)(ws + 16777216);             // 16384x1536
  _Float16* attn_h = (_Float16*)(ws + 16777216 + 50331648);  // 16384x512
  _Float16* wqkvT  = (_Float16*)(ws + 83886080);             // 1536x512
  _Float16* woutT  = (_Float16*)(ws + 85458944);             // 512x512

  // 0) conversions (bandwidth-trivial)
  cvt_f16_kernel<<<dim3(16384 * 512 / 8 / 256), 256, 0, stream>>>(
      x, x_h, 16384 * 512);
  transpose_f16_kernel<<<dim3(1536 / 32, 512 / 32), 256, 0, stream>>>(
      w_qkv, wqkvT, 512, 1536);
  transpose_f16_kernel<<<dim3(512 / 32, 512 / 32), 256, 0, stream>>>(
      w_out, woutT, 512, 512);

  // 1) qkv = x @ W_qkv : [16384x512] x [512x1536]
  wmma_gemm_f16<true><<<dim3(1536 / 128, 16384 / 128), 256, 0, stream>>>(
      x_h, wqkvT, qkv_h, 512, 512, 512, 1536);

  // 2) flash attention per (bm, head, 128-row slab)
  flash_attn_kernel<<<dim3(32 * 8 * 4), 256, 0, stream>>>(qkv_h, attn_h);

  // 3) out = attn @ W_out : [16384x512] x [512x512]
  wmma_gemm_f16<false><<<dim3(512 / 128, 16384 / 128), 256, 0, stream>>>(
      attn_h, woutT, out, 512, 512, 512, 512);
}
